// spec_18399639896890
// MI455X (gfx1250) — matrix operand order repaired and checked by reading, not compiled, not run on hardware
//
#include <hip/hip_runtime.h>
#include <hip/hip_bf16.h>
#include <hip/hip_fp16.h>

#define Nv 8192
#define Ev 49152

typedef __attribute__((ext_vector_type(16))) _Float16 v16h;
typedef __attribute__((ext_vector_type(8)))  _Float16 v8h;
typedef __attribute__((ext_vector_type(8)))  float    v8f;

// ---------------------------------------------------------------------------
// Transpose + convert: ut[j][i] = (f16) u[i][j]; optionally also emit the
// row-major f16 copy u16[i][j] (free: u is being streamed anyway).
// ---------------------------------------------------------------------------
__global__ __launch_bounds__(1024) void k_transpose_f16(
    const float* __restrict__ u, _Float16* __restrict__ ut,
    _Float16* __restrict__ u16) {
  __shared__ float tile[32][33];
  int tx = threadIdx.x, ty = threadIdx.y;
  int x = blockIdx.x * 32 + tx;
  int y = blockIdx.y * 32 + ty;
  float v = u[(size_t)y * Nv + x];
  tile[ty][tx] = v;
  if (u16) u16[(size_t)y * Nv + x] = (_Float16)v;
  __syncthreads();
  int ox = blockIdx.y * 32 + tx;
  int oy = blockIdx.x * 32 + ty;
  ut[(size_t)oy * Nv + ox] = (_Float16)tile[tx][ty];
}

// ---------------------------------------------------------------------------
// Encode: s0 = lamda @ W_enc + b_enc packed transposed into P1 rows 0..6,
// plus x^T into P1 rows 7..9 (P1 = stage-1 RHS panel, f16 [16, N]).
// ---------------------------------------------------------------------------
__global__ void k_encode_pack1(const float* __restrict__ lamda,
                               const float* __restrict__ We,
                               const float* __restrict__ be,
                               const float* __restrict__ x,
                               _Float16* __restrict__ P1) {
  int n = blockIdx.x * blockDim.x + threadIdx.x;
  if (n >= Nv) return;
  float l = lamda[n];
#pragma unroll
  for (int d = 0; d < 7; ++d)
    P1[(size_t)d * Nv + n] = (_Float16)(l * We[d] + be[d]);
#pragma unroll
  for (int c = 0; c < 3; ++c)
    P1[(size_t)(7 + c) * Nv + n] = (_Float16)x[n * 3 + c];
}

// ---------------------------------------------------------------------------
// WMMA GEMM:  C[M=8192, ncols] = A[M,K] @ B   (+bias)
//   A: row-major f16 (or fp32 converted in-register), leading dim lda
//   B: stored transposed as f16 panel Bt [NT*16, K] row-major
//   C: fp32 row-major, pitch ldc, column offset colOff, store cols < ncols
// Block = 64 threads = 2 wave32; wave w owns rows blockIdx.x*32 + w*16 .. +15
// (256 blocks over M=8192 -> more independent memory streams for 23.3 TB/s).
// Fragment layouts per CDNA5 ISA §7.12.2 (wave32):
//   A 16x32 f16 : lane l -> M=l&15 ; elems 0..7 = K kb+(l>>4)*8.. , 8..15 = +16
//   B 32x16 f16 : lane l -> N=l&15 ; elems 0..7 = K kb+(l>>4)*8.. , 8..15 = +16
//   C 16x16 f32 : lane l -> N=l&15 ; VGPR i -> M = (l>>4)*8 + i
// ---------------------------------------------------------------------------
template <bool AF32, int NT>
__global__ __launch_bounds__(64) void k_gemm(
    const void* __restrict__ Ap, int lda,
    const _Float16* __restrict__ Bt, int ldb,
    float* __restrict__ C, int ldc, int colOff, int ncols,
    const float* __restrict__ bias, int K) {
  const int lane = threadIdx.x & 31;
  const int wave = threadIdx.x >> 5;
  const int r    = lane & 15;
  const int hi   = lane >> 4;
  const int m0   = blockIdx.x * 32 + wave * 16;

  v8f acc[NT];
#pragma unroll
  for (int t = 0; t < NT; ++t)
#pragma unroll
    for (int i = 0; i < 8; ++i) acc[t][i] = 0.0f;

  for (int kb = 0; kb < K; kb += 32) {
    v16h a;
    if (AF32) {
      const float* Ar = (const float*)Ap + (size_t)(m0 + r) * lda + kb + hi * 8;
      v8f f0 = *(const v8f*)(Ar);
      v8f f1 = *(const v8f*)(Ar + 16);
#pragma unroll
      for (int i = 0; i < 8; ++i) { a[i] = (_Float16)f0[i]; a[i + 8] = (_Float16)f1[i]; }
    } else {
      const _Float16* Ar = (const _Float16*)Ap + (size_t)(m0 + r) * lda + kb + hi * 8;
      v8h h0 = *(const v8h*)(Ar);
      v8h h1 = *(const v8h*)(Ar + 16);
#pragma unroll
      for (int i = 0; i < 8; ++i) { a[i] = h0[i]; a[i + 8] = h1[i]; }
    }
#pragma unroll
    for (int t = 0; t < NT; ++t) {
      const _Float16* Br = Bt + (size_t)(t * 16 + r) * ldb + kb + hi * 8;
      v8h g0 = *(const v8h*)(Br);
      v8h g1 = *(const v8h*)(Br + 16);
      v16h b;
#pragma unroll
      for (int i = 0; i < 8; ++i) { b[i] = g0[i]; b[i + 8] = g1[i]; }
      acc[t] = __builtin_amdgcn_wmma_f32_16x16x32_f16(
          false, a, false, b, (short)0, acc[t], false, false);
    }
  }

  const int rowb = m0 + hi * 8;
#pragma unroll
  for (int t = 0; t < NT; ++t) {
    int col = t * 16 + r;
    if (col < ncols) {
      float bv = bias ? bias[col] : 0.0f;
#pragma unroll
      for (int i = 0; i < 8; ++i)
        C[(size_t)(rowb + i) * ldc + colOff + col] = acc[t][i] + bv;
    }
  }
}

// ---------------------------------------------------------------------------
// EdgeConv: m = [x_i, x_j - x_i] @ W + b ; segment-max at dst via monotone
// uint encoding + atomicMax. One edge per 128-thread block.
// ---------------------------------------------------------------------------
__device__ __forceinline__ unsigned encMax(float v) {
  unsigned u = __float_as_uint(v);
  return (u & 0x80000000u) ? ~u : (u | 0x80000000u);
}

__global__ __launch_bounds__(128) void k_econv(
    const float* __restrict__ xin, int din,
    const float* __restrict__ W, const float* __restrict__ bias, int h,
    const int* __restrict__ ei, float* __restrict__ out, int ldc) {
  __shared__ float sxi[128];
  __shared__ float sdx[128];
  int e   = blockIdx.x;
  int tid = threadIdx.x;
  int src = ei[e];
  int dst = ei[Ev + e];
  if (tid < din) {
    float a = xin[(size_t)dst * din + tid];
    float b = xin[(size_t)src * din + tid];
    sxi[tid] = a;
    sdx[tid] = b - a;
  }
  __syncthreads();
  if (tid < h) {
    float acc = bias[tid];
    for (int f = 0; f < din; ++f)
      acc += sxi[f] * W[f * h + tid] + sdx[f] * W[(din + f) * h + tid];
    atomicMax((unsigned*)(out + (size_t)dst * ldc + tid), encMax(acc));
  }
}

__global__ void k_init_neg(float* out, int ldc, int h) {
  int t = blockIdx.x * blockDim.x + threadIdx.x;
  if (t >= Nv * h) return;
  int n = t / h, c = t % h;
  ((unsigned*)out)[(size_t)n * ldc + c] = 0x007FFFFFu;  // encMax(-inf)
}

__global__ void k_fixup(float* out, int ldc, int h) {
  int t = blockIdx.x * blockDim.x + threadIdx.x;
  if (t >= Nv * h) return;
  int n = t / h, c = t % h;
  unsigned k = ((unsigned*)out)[(size_t)n * ldc + c];
  unsigned u = (k & 0x80000000u) ? (k & 0x7FFFFFFFu) : ~k;
  float v = __uint_as_float(u);
  out[(size_t)n * ldc + c] = __builtin_isfinite(v) ? v : 0.0f;
}

// ---------------------------------------------------------------------------
// Pack fp32 [N, D] row-major -> f16 panel rows [rowOff .. rowOff+D), [*, N]
// ---------------------------------------------------------------------------
__global__ void k_pack_t(const float* __restrict__ src, int ld, int d,
                         _Float16* __restrict__ panel, int rowOff) {
  int t = blockIdx.x * blockDim.x + threadIdx.x;
  if (t >= d * Nv) return;
  int dd = t / Nv, n = t % Nv;
  panel[(size_t)(rowOff + dd) * Nv + n] = (_Float16)src[(size_t)n * ld + dd];
}

__global__ void k_convert_h(const float* __restrict__ src,
                            _Float16* __restrict__ dst, int count) {
  int t = blockIdx.x * blockDim.x + threadIdx.x;
  if (t < count) dst[t] = (_Float16)src[t];
}

__global__ void k_pack_wdec(const float* __restrict__ W, _Float16* __restrict__ Wt) {
  int t = blockIdx.x * blockDim.x + threadIdx.x;
  if (t >= 32 * 128) return;
  int o = t / 128, kk = t % 128;
  Wt[o * 128 + kk] = (_Float16)W[kk * 32 + o];
}

// ---------------------------------------------------------------------------
extern "C" void kernel_launch(void* const* d_in, const int* in_sizes, int n_in,
                              void* d_out, int out_size, void* d_ws, size_t ws_size,
                              hipStream_t stream) {
  const float* x     = (const float*)d_in[0];
  const int*   ei    = (const int*)d_in[1];
  const float* u     = (const float*)d_in[2];
  const float* lamda = (const float*)d_in[3];
  const float* W_enc = (const float*)d_in[4];
  const float* b_enc = (const float*)d_in[5];
  const float* W1 = (const float*)d_in[6];  const float* b1 = (const float*)d_in[7];
  const float* W2 = (const float*)d_in[8];  const float* b2 = (const float*)d_in[9];
  const float* W3 = (const float*)d_in[10]; const float* b3 = (const float*)d_in[11];
  const float* W4 = (const float*)d_in[12]; const float* b4 = (const float*)d_in[13];
  const float* Wd = (const float*)d_in[14]; const float* bd = (const float*)d_in[15];
  float* out = (float*)d_out;

  char* w = (char*)d_ws;
  auto alloc = [&](size_t bytes) -> char* {
    char* p = w;
    w += (bytes + 255) & ~(size_t)255;
    return p;
  };
  _Float16* ut16 = (_Float16*)alloc((size_t)Nv * Nv * 2);   // 128 MB f16 u^T
  _Float16* P1   = (_Float16*)alloc((size_t)16 * Nv * 2);   // stage RHS panels
  _Float16* P2   = (_Float16*)alloc((size_t)64 * Nv * 2);
  _Float16* P3   = (_Float16*)alloc((size_t)96 * Nv * 2);
  _Float16* P4   = (_Float16*)alloc((size_t)96 * Nv * 2);
  float* s1  = (float*)alloc((size_t)Nv * 10 * 4);
  float* s2  = (float*)alloc((size_t)Nv * 49 * 4);
  float* s3  = (float*)alloc((size_t)Nv * 91 * 4);
  float* x1c = (float*)alloc((size_t)Nv * 39 * 4);
  float* x2c = (float*)alloc((size_t)Nv * 42 * 4);
  float* x3c = (float*)alloc((size_t)Nv * 113 * 4);
  float* x4c = (float*)alloc((size_t)Nv * 128 * 4);
  _Float16* x4h = (_Float16*)alloc((size_t)Nv * 128 * 2);
  _Float16* wdt = (_Float16*)alloc((size_t)32 * 128 * 2);
  size_t needSmall = (size_t)(w - (char*)d_ws);
  _Float16* u16 = (_Float16*)alloc((size_t)Nv * Nv * 2);    // optional 128 MB
  size_t needBig = (size_t)(w - (char*)d_ws);
  const bool useU16 = (ws_size >= needBig);
  (void)needSmall;

  const dim3 GG(256), GB(64);  // 256 blocks x 2 waves for all WMMA GEMMs

  // u^T (and optionally row-major u) in f16
  k_transpose_f16<<<dim3(Nv / 32, Nv / 32), dim3(32, 32), 0, stream>>>(
      u, ut16, useU16 ? u16 : nullptr);

  // zero all RHS panels (covers zero-padded tile rows)
  size_t panelBytes = (size_t)((char*)P4 + (size_t)96 * Nv * 2 - (char*)P1);
  hipMemsetAsync(P1, 0, panelBytes, stream);

  k_encode_pack1<<<Nv / 256, 256, 0, stream>>>(lamda, W_enc, b_enc, x, P1);

  // -inf (encoded) init of edge-conv output columns
  k_init_neg<<<(Nv * 32 + 255) / 256, 256, 0, stream>>>(x1c, 39, 32);
  k_init_neg<<<(Nv * 32 + 255) / 256, 256, 0, stream>>>(x2c, 42, 32);
  k_init_neg<<<(Nv * 64 + 255) / 256, 256, 0, stream>>>(x3c, 113, 64);
  k_init_neg<<<(Nv * 37 + 255) / 256, 256, 0, stream>>>(x4c, 128, 37);

  // ---- stage 1 : s1 = u^T @ [s0|x] ; x1 = [econv1(x) , u @ s0]
  k_gemm<false, 1><<<GG, GB, 0, stream>>>(ut16, Nv, P1, Nv, s1, 10, 0, 10, nullptr, Nv);
  if (useU16) k_gemm<false, 1><<<GG, GB, 0, stream>>>(u16, Nv, P1, Nv, x1c, 39, 32, 7, nullptr, Nv);
  else        k_gemm<true,  1><<<GG, GB, 0, stream>>>(u,   Nv, P1, Nv, x1c, 39, 32, 7, nullptr, Nv);
  k_econv<<<Ev, 128, 0, stream>>>(x, 3, W1, b1, 32, ei, x1c, 39);
  k_fixup<<<(Nv * 32 + 255) / 256, 256, 0, stream>>>(x1c, 39, 32);

  // ---- stage 2 : s2 = u^T @ [s1|x1] ; x2 = [econv2(x1) , u @ s1]
  k_pack_t<<<(10 * Nv + 255) / 256, 256, 0, stream>>>(s1, 10, 10, P2, 0);
  k_pack_t<<<(39 * Nv + 255) / 256, 256, 0, stream>>>(x1c, 39, 39, P2, 10);
  k_gemm<false, 4><<<GG, GB, 0, stream>>>(ut16, Nv, P2, Nv, s2, 49, 0, 49, nullptr, Nv);
  if (useU16) k_gemm<false, 1><<<GG, GB, 0, stream>>>(u16, Nv, P2, Nv, x2c, 42, 32, 10, nullptr, Nv);
  else        k_gemm<true,  1><<<GG, GB, 0, stream>>>(u,   Nv, P2, Nv, x2c, 42, 32, 10, nullptr, Nv);
  k_econv<<<Ev, 128, 0, stream>>>(x1c, 39, W2, b2, 32, ei, x2c, 42);
  k_fixup<<<(Nv * 32 + 255) / 256, 256, 0, stream>>>(x2c, 42, 32);

  // ---- stage 3 : s3 = u^T @ [s2|x2] ; x3 = [econv3(x2) , u @ s2]
  k_pack_t<<<(49 * Nv + 255) / 256, 256, 0, stream>>>(s2, 49, 49, P3, 0);
  k_pack_t<<<(42 * Nv + 255) / 256, 256, 0, stream>>>(x2c, 42, 42, P3, 49);
  k_gemm<false, 6><<<GG, GB, 0, stream>>>(ut16, Nv, P3, Nv, s3, 91, 0, 91, nullptr, Nv);
  if (useU16) k_gemm<false, 4><<<GG, GB, 0, stream>>>(u16, Nv, P3, Nv, x3c, 113, 64, 49, nullptr, Nv);
  else        k_gemm<true,  4><<<GG, GB, 0, stream>>>(u,   Nv, P3, Nv, x3c, 113, 64, 49, nullptr, Nv);
  k_econv<<<Ev, 128, 0, stream>>>(x2c, 42, W3, b3, 64, ei, x3c, 113);
  k_fixup<<<(Nv * 64 + 255) / 256, 256, 0, stream>>>(x3c, 113, 64);

  // ---- stage 4 : x4 = [econv4(x3) , u @ s3]
  k_pack_t<<<(91 * Nv + 255) / 256, 256, 0, stream>>>(s3, 91, 91, P4, 0);
  if (useU16) k_gemm<false, 6><<<GG, GB, 0, stream>>>(u16, Nv, P4, Nv, x4c, 128, 37, 91, nullptr, Nv);
  else        k_gemm<true,  6><<<GG, GB, 0, stream>>>(u,   Nv, P4, Nv, x4c, 128, 37, 91, nullptr, Nv);
  k_econv<<<Ev, 128, 0, stream>>>(x3c, 113, W4, b4, 37, ei, x4c, 128);
  k_fixup<<<(Nv * 37 + 255) / 256, 256, 0, stream>>>(x4c, 128, 37);

  // ---- decode : out = x4 @ W_dec + b_dec   (K = 128, via WMMA)
  k_convert_h<<<(Nv * 128 + 255) / 256, 256, 0, stream>>>(x4c, x4h, Nv * 128);
  k_pack_wdec<<<(32 * 128 + 255) / 256, 256, 0, stream>>>(Wd, wdt);
  k_gemm<false, 2><<<GG, GB, 0, stream>>>(x4h, 128, wdt, 128, out, 32, 0, 32, bd, 128);
}
